// HyperbolicLinear_36945308680257
// MI455X (gfx1250) — matrix operand order repaired and checked by reading, not compiled, not run on hardware
//
#include <hip/hip_runtime.h>
#include <hip/hip_bf16.h>
#include <math.h>

typedef __attribute__((ext_vector_type(16))) _Float16 v16h;
typedef __attribute__((ext_vector_type(8)))  _Float16 v8h;
typedef __attribute__((ext_vector_type(8)))  float    v8f;

#define EPSV 1e-7f
#define MAXN (1.0f - 1e-5f)
#define NROWS 262144
#define DIM 256
#define RPB 32            // rows per block
#define LDSW 264          // padded LDS row width in halves (256 + 8)

// ---- prep: W (f32, row-major [OUT][IN]) -> f16 in workspace ----
__global__ void convert_w_f16(const float* __restrict__ W, _Float16* __restrict__ Wh) {
    int i = blockIdx.x * blockDim.x + threadIdx.x;
    Wh[i] = (_Float16)W[i];
}

// ---- prep: ||b||^2 ----
__global__ void bnorm2_kernel(const float* __restrict__ b, float* __restrict__ outp) {
    __shared__ float red[8];
    int t = threadIdx.x;
    float v = b[t];
    float p = v * v;
    #pragma unroll
    for (int off = 16; off >= 1; off >>= 1) p += __shfl_xor(p, off, 32);
    if ((t & 31) == 0) red[t >> 5] = p;
    __syncthreads();
    if (t == 0) {
        float s = 0.f;
        #pragma unroll
        for (int i = 0; i < 8; ++i) s += red[i];
        *outp = s;
    }
}

// ---- fused hyperbolic linear: GEMM (WMMA f16) + row reductions + epilogue ----
__launch_bounds__(256)
__global__ void hyp_linear_kernel(const float* __restrict__ x,
                                  const _Float16* __restrict__ Wh,
                                  const float* __restrict__ b,
                                  const float* __restrict__ b2p,
                                  float* __restrict__ out) {
    __shared__ _Float16 xs[RPB * LDSW];
    __shared__ float s_x2[RPB], s_mx2[RPB], s_mxb[RPB], s_c1[RPB], s_c2[RPB];

    const int t    = threadIdx.x;
    const int lane = t & 31;
    const int wave = t >> 5;
    const long long m0 = (long long)blockIdx.x * RPB;

    if (t < RPB) { s_x2[t] = 0.f; s_mx2[t] = 0.f; s_mxb[t] = 0.f; }
    __syncthreads();

    // ---- stage x tile [32 x 256] into LDS as f16; accumulate sum(x^2) per row ----
    {
        const int row = t >> 3;           // 32 rows, 8 threads per row
        const int seg = t & 7;
        const float* xr = x + (m0 + row) * DIM;
        _Float16* ls = xs + row * LDSW;
        float acc = 0.f;
        #pragma unroll
        for (int i = 0; i < 8; ++i) {
            const int c = seg * 4 + i * 32;        // coalesced float4 across the row group
            float4 v = *(const float4*)(xr + c);
            acc += v.x * v.x + v.y * v.y + v.z * v.z + v.w * v.w;
            ls[c + 0] = (_Float16)v.x;
            ls[c + 1] = (_Float16)v.y;
            ls[c + 2] = (_Float16)v.z;
            ls[c + 3] = (_Float16)v.w;
        }
        atomicAdd(&s_x2[row], acc);
    }
    __syncthreads();

    // ---- WMMA GEMM: wave tile = 16 rows x 64 cols (4 accumulators) ----
    const int waveM = wave >> 2;      // 0..1
    const int waveN = wave & 3;       // 0..3
    const int col0  = waveN * 64;
    const int lrow  = lane & 15;
    const int hi    = lane >> 4;      // half-wave select
    const int arow  = waveM * 16 + lrow;

    v8f acc[4] = {};
    for (int k0 = 0; k0 < DIM; k0 += 32) {
        // A fragment: 16x32 f16, lane = row%16, K halves {0..15} (lo half-wave)
        // or {16..31} (hi half-wave), contiguous order matching B.
        v16h a;
        {
            const _Float16* ap = xs + arow * LDSW + k0 + hi * 16;
            v8h lo  = *(const v8h*)(ap);
            v8h hi8 = *(const v8h*)(ap + 8);
            #pragma unroll
            for (int i = 0; i < 8; ++i) { a[i] = lo[i]; a[i + 8] = hi8[i]; }
        }
        #pragma unroll
        for (int tt = 0; tt < 4; ++tt) {
            const int col = col0 + tt * 16 + lrow;
            // B fragment: 32x16, lane = col, 16 contiguous K halves from row-major W
            v16h bf = *(const v16h*)(Wh + col * DIM + k0 + hi * 16);
            acc[tt] = __builtin_amdgcn_wmma_f32_16x16x32_f16(
                false, a, false, bf, (short)0, acc[tt], false, false);
        }
    }

    // ---- per-row reductions of sum(mx^2) and sum(mx*b) from accumulators ----
    float bv[4];
    #pragma unroll
    for (int tt = 0; tt < 4; ++tt) bv[tt] = b[col0 + tt * 16 + lrow];

    #pragma unroll
    for (int v = 0; v < 8; ++v) {
        float p2 = 0.f, pb = 0.f;
        #pragma unroll
        for (int tt = 0; tt < 4; ++tt) {
            const float m = acc[tt][v];
            p2 += m * m;
            pb += m * bv[tt];
        }
        #pragma unroll
        for (int off = 1; off <= 8; off <<= 1) {   // reduce within 16-lane half-wave
            p2 += __shfl_xor(p2, off, 32);
            pb += __shfl_xor(pb, off, 32);
        }
        if (lrow == 0) {
            const int rl = waveM * 16 + hi * 8 + v;   // C/D layout: row = v + 8*hi
            atomicAdd(&s_mx2[rl], p2);
            atomicAdd(&s_mxb[rl], pb);
        }
    }
    __syncthreads();

    // ---- per-row scalar chain: mobius_matvec tail + transport + expmap collapse ----
    if (t < RPB) {
        const float b2  = *b2p;
        const float bn  = sqrtf(b2);
        const float x2  = s_x2[t];
        const float mx2 = s_mx2[t];
        const float mxb = s_mxb[t];

        const float xn  = fmaxf(sqrtf(x2), EPSV);
        const float mxn = fmaxf(sqrtf(mx2), EPSV);
        const float xc  = fminf(xn, 1.0f - 1e-7f);
        const float art = 0.5f * (log1pf(xc) - log1pf(-xc));
        const float r   = tanhf(mxn / xn * art);
        float s = r / mxn;                       // res = s * mx
        const float resn = fabsf(s) * sqrtf(mx2);
        if (resn > MAXN) s *= MAXN / fmaxf(resn, EPSV);   // _proj(res)

        const float res2 = s * s * mx2;
        const float omr  = 1.0f - res2;
        // pt = omr*b ; second = tanh(0.5*lam*||pt||) * pt/||pt|| = alpha * b
        const float vn    = fmaxf(fabsf(omr) * bn, EPSV);
        const float lam   = 2.0f / fmaxf(omr, EPSV);
        const float alpha = tanhf(0.5f * lam * vn) * omr / vn;

        // mobius_add(res, alpha*b)
        const float y2 = alpha * alpha * b2;
        const float xy = s * alpha * mxb;
        const float An = 1.0f + 2.0f * xy + y2;
        const float dn = fmaxf(1.0f + 2.0f * xy + res2 * y2, EPSV);
        float c1 = An * s / dn;       // coefficient of mx
        float c2 = omr * alpha / dn;  // coefficient of b

        // final _proj on c1*mx + c2*b
        float fn = sqrtf(fmaxf(c1 * c1 * mx2 + 2.f * c1 * c2 * mxb + c2 * c2 * b2, 0.f));
        fn = fmaxf(fn, EPSV);
        if (fn > MAXN) { const float k = MAXN / fn; c1 *= k; c2 *= k; }
        s_c1[t] = c1; s_c2[t] = c2;
    }
    __syncthreads();

    // ---- epilogue: out = c1*mx + c2*b, straight from accumulators ----
    #pragma unroll
    for (int v = 0; v < 8; ++v) {
        const int rl = waveM * 16 + hi * 8 + v;
        const float c1 = s_c1[rl];
        const float c2 = s_c2[rl];
        const long long grow = m0 + rl;
        #pragma unroll
        for (int tt = 0; tt < 4; ++tt) {
            const int col = col0 + tt * 16 + lrow;
            out[grow * DIM + col] = c1 * acc[tt][v] + c2 * bv[tt];
        }
    }
}

extern "C" void kernel_launch(void* const* d_in, const int* in_sizes, int n_in,
                              void* d_out, int out_size, void* d_ws, size_t ws_size,
                              hipStream_t stream) {
    const float* x = (const float*)d_in[0];   // [262144, 256] f32
    const float* W = (const float*)d_in[1];   // [256, 256] f32
    const float* b = (const float*)d_in[2];   // [256] f32
    float* out = (float*)d_out;

    _Float16* Wh = (_Float16*)d_ws;                                   // 128 KB
    float* b2 = (float*)((char*)d_ws + DIM * DIM * sizeof(_Float16)); // scalar

    convert_w_f16<<<DIM * DIM / 256, 256, 0, stream>>>(W, Wh);
    bnorm2_kernel<<<1, 256, 0, stream>>>(b, b2);
    hyp_linear_kernel<<<NROWS / RPB, 256, 0, stream>>>(x, Wh, b, b2, out);
}
